// EncDecModel_22617297781039
// MI455X (gfx1250) — compile-verified, matrix operand order checked by reading, not run on hardware
//
#include <hip/hip_runtime.h>

#define B_    64
#define TENC_ 40
#define TCAP_ 20
#define DIN_  2048
#define HE_   512
#define HD_   1024
#define E_    512
#define C_    512
#define V_    10000

typedef __bf16 bf16_t;
typedef __attribute__((ext_vector_type(16))) __bf16        v16bf;
typedef __attribute__((ext_vector_type(8)))  float         v8f;
typedef __attribute__((ext_vector_type(8)))  unsigned short u16x8;
typedef __attribute__((ext_vector_type(16))) unsigned short u16x16;

// ---------------------------------------------------------------------------
// Tile loader for the CDNA5 16-bit 16x32 A/B WMMA layout.
// Lane l (l15 = l&15, hi = l>>4) holds row (tileRow + l15), K chunks
// {khalf..khalf+7} and {khalf+16..khalf+23} with khalf = hi*8.
// Two global_load_b128 per lane.
// ---------------------------------------------------------------------------
__device__ __forceinline__ v16bf load_tile_b128x2(const bf16_t* p) {
  u16x8 lo = *reinterpret_cast<const u16x8*>(p);
  u16x8 hi = *reinterpret_cast<const u16x8*>(p + 16);
  u16x16 t;
#pragma unroll
  for (int i = 0; i < 8; ++i) { t[i] = lo[i]; t[i + 8] = hi[i]; }
  return __builtin_bit_cast(v16bf, t);
}

// ---------------------------------------------------------------------------
// Generic bf16 WMMA GEMM:  C[M,N] = A[M,K] * B[N,K]^T (+Cinit)(+bias)(tanh?)
// Block = 256 threads = 8 wave32s. Block tile = 64M x (32*NT)N.
// Wave w: M tile 16*(w&3); N tiles (w>>2)*(16*NT) + {0,16,..} -> NT WMMAs per
// A-tile load. Ragged N handled by CLAMPING the B row (no exec divergence).
// Software-pipelined: iteration k+1's tile loads are issued before iteration
// k's WMMAs so the matrix pipe overlaps with VMEM (partial loadcnt waits).
// M must be a multiple of 64 (true for all call sites).
// ---------------------------------------------------------------------------
template <int NT>
__global__ __launch_bounds__(256)
void gemm_bf16_wmma(const bf16_t* __restrict__ A, int lda,
                    const bf16_t* __restrict__ Bm, int ldb,
                    const float* Cinit, int ldci,
                    const float* __restrict__ bias,
                    float* C, int ldc,
                    bf16_t* __restrict__ Cbf, int ldcbf,
                    int N, int K, int act)
{
  const int wave  = threadIdx.x >> 5;
  const int lane  = threadIdx.x & 31;
  const int l15   = lane & 15;
  const int khalf = (lane >> 4) * 8;

  const int mBase = blockIdx.y * 64 + 16 * (wave & 3);
  const int nBase = blockIdx.x * (32 * NT) + (wave >> 2) * (16 * NT);

  const int aRow = mBase + l15;
  const bf16_t* aPtr = A + (long)aRow * lda + khalf;

  const bf16_t* bPtr[NT];
#pragma unroll
  for (int j = 0; j < NT; ++j) {
    const int r  = nBase + 16 * j + l15;
    const int rc = r < N ? r : N - 1;     // clamp instead of zero-fill
    bPtr[j] = Bm + (long)rc * ldb + khalf;
  }

  const v8f vzero = {0.f, 0.f, 0.f, 0.f, 0.f, 0.f, 0.f, 0.f};
  v8f acc[NT];
#pragma unroll
  for (int j = 0; j < NT; ++j) acc[j] = vzero;

  // ---- software pipeline: prologue loads for k = 0 ----
  v16bf aCur = load_tile_b128x2(aPtr);
  v16bf bCur[NT];
#pragma unroll
  for (int j = 0; j < NT; ++j) bCur[j] = load_tile_b128x2(bPtr[j]);

#pragma unroll 2
  for (int k = 32; k < K; k += 32) {
    // issue next iteration's loads first (one full iteration of latency hide)
    const v16bf aNxt = load_tile_b128x2(aPtr + k);
    v16bf bNxt[NT];
#pragma unroll
    for (int j = 0; j < NT; ++j) bNxt[j] = load_tile_b128x2(bPtr[j] + k);
    if (k + 96 < K) {                     // stream next K block into GL2
      __builtin_prefetch(aPtr + k + 96, 0, 3);
#pragma unroll
      for (int j = 0; j < NT; ++j) __builtin_prefetch(bPtr[j] + k + 96, 0, 3);
    }
    // consume current tiles while next loads are in flight
#pragma unroll
    for (int j = 0; j < NT; ++j)
      acc[j] = __builtin_amdgcn_wmma_f32_16x16x32_bf16(false, aCur, false, bCur[j],
                                                       (short)0, acc[j],
                                                       false, false);
    aCur = aNxt;
#pragma unroll
    for (int j = 0; j < NT; ++j) bCur[j] = bNxt[j];
  }
  // ---- epilogue of the pipeline: last tile set ----
#pragma unroll
  for (int j = 0; j < NT; ++j)
    acc[j] = __builtin_amdgcn_wmma_f32_16x16x32_bf16(false, aCur, false, bCur[j],
                                                     (short)0, acc[j],
                                                     false, false);

  // C/D layout: VGPR g, lanes 0-15 -> M = g, lanes 16-31 -> M = 8+g; N = l15.
  const int mRow = mBase + 8 * (lane >> 4);
#pragma unroll
  for (int j = 0; j < NT; ++j) {
    const int n = nBase + 16 * j + l15;
    if (n >= N) continue;
#pragma unroll
    for (int g = 0; g < 8; ++g) {
      const int m = mRow + g;
      float r = acc[j][g];
      if (Cinit) r += Cinit[(long)m * ldci + n];
      if (bias)  r += bias[n];
      if (act == 1) r = tanhf(r);
      C[(long)m * ldc + n] = r;
      if (Cbf) Cbf[(long)m * ldcbf + n] = (bf16_t)r;
    }
  }
}

// ---------------------------------------------------------------------------
// Fused LSTM cell: gates[B,4H] (pre-activation, bias already folded in) + c.
// Gate order i,f,g,o (JAX split order). Writes fp32 h (optional) and up to two
// bf16 mirrors (recurrence input / GEMM-A copies).
// ---------------------------------------------------------------------------
__device__ __forceinline__ float sigm_(float x) { return 1.f / (1.f + expf(-x)); }

__global__ void lstm_cell_k(const float* __restrict__ gates, float* __restrict__ c,
                            float* hf, long hf_rs,
                            bf16_t* __restrict__ hb1, long hb1_rs,
                            bf16_t* __restrict__ hb2, long hb2_rs, int H)
{
  const int idx = blockIdx.x * blockDim.x + threadIdx.x;
  if (idx >= B_ * H) return;
  const int b = idx / H;
  const int j = idx - b * H;
  const float* g = gates + (long)b * 4 * H;
  const float gi = g[j], gf = g[H + j], gg = g[2 * H + j], go = g[3 * H + j];
  const float cn = sigm_(gf) * c[idx] + sigm_(gi) * tanhf(gg);
  const float hn = sigm_(go) * tanhf(cn);
  c[idx] = cn;
  if (hf)  hf[(long)b * hf_rs + j] = hn;
  hb1[(long)b * hb1_rs + j] = (bf16_t)hn;
  if (hb2) hb2[(long)b * hb2_rs + j] = (bf16_t)hn;
}

// ---------------------------------------------------------------------------
// Bahdanau attention for one decoder step. One block per batch row.
// scores[t] = sum_c tanh(v1[b,c]+v2[b,t,c]) * Wa3[c]; softmax over t;
// ctx[b,:] = sum_t a[t] * enc[b,t,:]  (written as bf16 for the next GEMM).
// Wave-shuffle + LDS reductions.
// ---------------------------------------------------------------------------
__global__ __launch_bounds__(256)
void attention_k(const float* __restrict__ v1, const float* __restrict__ v2,
                 const float* __restrict__ Wa3, const float* __restrict__ enc,
                 bf16_t* __restrict__ ctx_bf)
{
  const int b    = blockIdx.x;
  const int wave = threadIdx.x >> 5;
  const int lane = threadIdx.x & 31;
  __shared__ float s_sc[TENC_];
  __shared__ float s_a[TENC_];

#pragma unroll
  for (int i = 0; i < TENC_ / 8; ++i) {
    const int t = wave + 8 * i;
    const float* v2p = v2 + ((long)b * TENC_ + t) * C_;
    const float* v1p = v1 + (long)b * C_;
    float s = 0.f;
    for (int cI = lane; cI < C_; cI += 32)
      s += tanhf(v1p[cI] + v2p[cI]) * Wa3[cI];
#pragma unroll
    for (int o = 16; o > 0; o >>= 1) s += __shfl_down(s, o);
    if (lane == 0) s_sc[t] = s;
  }
  __syncthreads();
  if (threadIdx.x == 0) {
    float mx = s_sc[0];
    for (int t = 1; t < TENC_; ++t) mx = fmaxf(mx, s_sc[t]);
    float sum = 0.f;
    for (int t = 0; t < TENC_; ++t) { float e = expf(s_sc[t] - mx); s_a[t] = e; sum += e; }
    const float inv = 1.f / sum;
    for (int t = 0; t < TENC_; ++t) s_a[t] *= inv;
  }
  __syncthreads();
  for (int d = threadIdx.x; d < 2 * HE_; d += 256) {
    float s = 0.f;
#pragma unroll 8
    for (int t = 0; t < TENC_; ++t)
      s += s_a[t] * enc[((long)b * TENC_ + t) * (2 * HE_) + d];
    ctx_bf[(long)b * (2 * HE_) + d] = (bf16_t)s;
  }
}

// ---------------------------------------------------------------------------
// Utility kernels.
// ---------------------------------------------------------------------------
__global__ void f32_to_bf16_k(const float* __restrict__ s, bf16_t* __restrict__ d, long n) {
  long i = (long)blockIdx.x * blockDim.x + threadIdx.x;
  const long stride = (long)gridDim.x * blockDim.x;
  for (; i < n; i += stride) d[i] = (bf16_t)s[i];
}

__global__ void zero_u32_k(unsigned* __restrict__ p, long n) {
  long i = (long)blockIdx.x * blockDim.x + threadIdx.x;
  const long stride = (long)gridDim.x * blockDim.x;
  for (; i < n; i += stride) p[i] = 0u;
}

// visual_features [B,T,DIN] -> bf16 A-matrix rows (t*B+b)
__global__ void pack_vis_k(const float* __restrict__ vis, bf16_t* __restrict__ out) {
  const long total = (long)TENC_ * B_ * DIN_;
  long i = (long)blockIdx.x * blockDim.x + threadIdx.x;
  const long stride = (long)gridDim.x * blockDim.x;
  for (; i < total; i += stride) {
    const int d = (int)(i % DIN_);
    const long r = i / DIN_;
    const int t = (int)(r / B_);
    const int b = (int)(r % B_);
    out[i] = (bf16_t)vis[((long)b * TENC_ + t) * DIN_ + d];
  }
}

// emb[caption_x] -> bf16 A-matrix rows (t*B+b)
__global__ void pack_emb_k(const float* __restrict__ emb, const int* __restrict__ cap,
                           bf16_t* __restrict__ out) {
  const long total = (long)TCAP_ * B_ * E_;
  long i = (long)blockIdx.x * blockDim.x + threadIdx.x;
  const long stride = (long)gridDim.x * blockDim.x;
  for (; i < total; i += stride) {
    const int e = (int)(i % E_);
    const long r = i / E_;
    const int t = (int)(r / B_);
    const int b = (int)(r % B_);
    const int tok = cap[b * TCAP_ + t];
    out[i] = (bf16_t)emb[(long)tok * E_ + e];
  }
}

// ---------------------------------------------------------------------------
// Host orchestration.
// ---------------------------------------------------------------------------
extern "C" void kernel_launch(void* const* d_in, const int* in_sizes, int n_in,
                              void* d_out, int out_size, void* d_ws, size_t ws_size,
                              hipStream_t stream)
{
  (void)in_sizes; (void)n_in; (void)out_size; (void)ws_size;

  const float* vis   = (const float*)d_in[0];
  const int*   cap   = (const int*)  d_in[1];
  const float* emb   = (const float*)d_in[2];
  const float* eWihF = (const float*)d_in[3];
  const float* eWhhF = (const float*)d_in[4];
  const float* ebF   = (const float*)d_in[5];
  const float* eWihB = (const float*)d_in[6];
  const float* eWhhB = (const float*)d_in[7];
  const float* ebB   = (const float*)d_in[8];
  const float* dWih  = (const float*)d_in[9];
  const float* dWhh  = (const float*)d_in[10];
  const float* db    = (const float*)d_in[11];
  const float* Wa1   = (const float*)d_in[12];
  const float* Wa2   = (const float*)d_in[13];
  const float* Wa3   = (const float*)d_in[14];
  const float* We2d  = (const float*)d_in[15];
  const float* Wout  = (const float*)d_in[16];
  const float* bout  = (const float*)d_in[17];

  char* ws = (char*)d_ws;
  size_t off = 0;
  auto alloc = [&](size_t bytes) -> char* {
    char* p = ws + off;
    off = (off + bytes + 255) & ~(size_t)255;
    return p;
  };

  // bf16 weight mirrors (all resident in L2 across the sequential loops)
  bf16_t* wWihF = (bf16_t*)alloc(2048L * 2048 * 2);
  bf16_t* wWihB = (bf16_t*)alloc(2048L * 2048 * 2);
  bf16_t* wWhhF = (bf16_t*)alloc(2048L * 512 * 2);
  bf16_t* wWhhB = (bf16_t*)alloc(2048L * 512 * 2);
  bf16_t* wDWih = (bf16_t*)alloc(4096L * 1024 * 2);
  bf16_t* wDWhh = (bf16_t*)alloc(4096L * 1024 * 2);
  bf16_t* wA1   = (bf16_t*)alloc(512L * 1024 * 2);
  bf16_t* wA2   = (bf16_t*)alloc(512L * 1024 * 2);
  bf16_t* wE2d  = (bf16_t*)alloc(512L * 1024 * 2);
  bf16_t* wO    = (bf16_t*)alloc(10000L * 1024 * 2);
  // activations / intermediates
  bf16_t* visbf  = (bf16_t*)alloc(2560L * 2048 * 2);
  float*  encXf  = (float*) alloc(2560L * 2048 * 4);
  float*  encXb  = (float*) alloc(2560L * 2048 * 4);
  float*  gatesF = (float*) alloc(64L * 2048 * 4);
  float*  gatesB = (float*) alloc(64L * 2048 * 4);
  bf16_t* hFbf   = (bf16_t*)alloc(64L * 512 * 2);
  bf16_t* hBbf   = (bf16_t*)alloc(64L * 512 * 2);
  float*  cF     = (float*) alloc(64L * 512 * 4);
  float*  cB     = (float*) alloc(64L * 512 * 4);
  float*  enc    = (float*) alloc(64L * 40 * 1024 * 4);
  bf16_t* encbf  = (bf16_t*)alloc(64L * 40 * 1024 * 2);
  float*  v2buf  = (float*) alloc(64L * 40 * 512 * 4);
  bf16_t* tokbf  = (bf16_t*)alloc(1280L * 512 * 2);
  float*  decX   = (float*) alloc(1280L * 4096 * 4);
  bf16_t* hDbf   = (bf16_t*)alloc(64L * 1024 * 2);
  float*  cD     = (float*) alloc(64L * 1024 * 4);
  float*  v1buf  = (float*) alloc(64L * 512 * 4);
  bf16_t* ctxbf  = (bf16_t*)alloc(64L * 1024 * 2);
  float*  ctxp   = (float*) alloc(64L * 512 * 4);
  bf16_t* ctx2bf = (bf16_t*)alloc(64L * 512 * 2);
  float*  gatesD = (float*) alloc(64L * 4096 * 4);
  bf16_t* hhbf   = (bf16_t*)alloc(1280L * 1024 * 2);

  auto cvt = [&](const float* s, bf16_t* d, long n) {
    f32_to_bf16_k<<<dim3(1024), dim3(256), 0, stream>>>(s, d, n);
  };
  auto zero = [&](void* p, long bytes) {
    zero_u32_k<<<dim3(64), dim3(256), 0, stream>>>((unsigned*)p, bytes / 4);
  };
  // nt=4: wide tiles for the big time-batched GEMMs (A-reuse x4).
  // nt=2: narrow tiles for the small sequential M=64 GEMMs (2x more blocks).
  auto gemm = [&](const bf16_t* A, int lda, const bf16_t* Bm, int ldb,
                  const float* Ci, int ldci, const float* bias,
                  float* Cc, int ldc, bf16_t* Cb, int ldcb,
                  int M, int N, int K, int act, int nt) {
    if (nt == 4) {
      dim3 grid((N + 127) / 128, M / 64);
      gemm_bf16_wmma<4><<<grid, dim3(256), 0, stream>>>(
          A, lda, Bm, ldb, Ci, ldci, bias, Cc, ldc, Cb, ldcb, N, K, act);
    } else {
      dim3 grid((N + 63) / 64, M / 64);
      gemm_bf16_wmma<2><<<grid, dim3(256), 0, stream>>>(
          A, lda, Bm, ldb, Ci, ldci, bias, Cc, ldc, Cb, ldcb, N, K, act);
    }
  };

  // --- weight conversion + input packing ---------------------------------
  cvt(eWihF, wWihF, 2048L * 2048);
  cvt(eWihB, wWihB, 2048L * 2048);
  cvt(eWhhF, wWhhF, 2048L * 512);
  cvt(eWhhB, wWhhB, 2048L * 512);
  cvt(dWih,  wDWih, 4096L * 1024);
  cvt(dWhh,  wDWhh, 4096L * 1024);
  cvt(Wa1,   wA1,   512L * 1024);
  cvt(Wa2,   wA2,   512L * 1024);
  cvt(We2d,  wE2d,  512L * 1024);
  cvt(Wout,  wO,    10000L * 1024);
  pack_vis_k<<<dim3(2048), dim3(256), 0, stream>>>(vis, visbf);
  pack_emb_k<<<dim3(512),  dim3(256), 0, stream>>>(emb, cap, tokbf);

  // --- zero recurrent state (fresh every call: deterministic) ------------
  zero(hFbf, 64L * 512 * 2);
  zero(hBbf, 64L * 512 * 2);
  zero(cF,   64L * 512 * 4);
  zero(cB,   64L * 512 * 4);
  zero(hDbf, 64L * 1024 * 2);
  zero(cD,   64L * 1024 * 4);

  // --- time-batched input projections (big GEMMs, bias folded in) --------
  gemm(visbf, 2048, wWihF, 2048, nullptr, 0, ebF, encXf, 2048, nullptr, 0,
       2560, 2048, 2048, 0, 4);
  gemm(visbf, 2048, wWihB, 2048, nullptr, 0, ebB, encXb, 2048, nullptr, 0,
       2560, 2048, 2048, 0, 4);
  gemm(tokbf, 512, wDWih, 1024, nullptr, 0, db, decX, 4096, nullptr, 0,
       1280, 4096, 512, 0, 4);

  // --- bidirectional encoder recurrence ----------------------------------
  const long encRS = 40L * 1024;
  for (int s = 0; s < TENC_; ++s) {
    const int tf = s, tb = TENC_ - 1 - s;
    gemm(hFbf, 512, wWhhF, 512, encXf + (long)tf * 64 * 2048, 2048, nullptr,
         gatesF, 2048, nullptr, 0, 64, 2048, 512, 0, 2);
    gemm(hBbf, 512, wWhhB, 512, encXb + (long)tb * 64 * 2048, 2048, nullptr,
         gatesB, 2048, nullptr, 0, 64, 2048, 512, 0, 2);
    lstm_cell_k<<<dim3(128), dim3(256), 0, stream>>>(
        gatesF, cF, enc + (long)tf * 1024, encRS, hFbf, 512,
        encbf + (long)tf * 1024, encRS, 512);
    lstm_cell_k<<<dim3(128), dim3(256), 0, stream>>>(
        gatesB, cB, enc + (long)tb * 1024 + 512, encRS, hBbf, 512,
        encbf + (long)tb * 1024 + 512, encRS, 512);
  }

  // --- attention keys: v2 = enc @ Wa2^T over all (b,t) -------------------
  gemm(encbf, 1024, wA2, 1024, nullptr, 0, nullptr, v2buf, 512, nullptr, 0,
       2560, 512, 1024, 0, 4);

  // --- teacher-forced decoder --------------------------------------------
  for (int t = 0; t < TCAP_; ++t) {
    // v1 = h @ Wa1^T
    gemm(hDbf, 1024, wA1, 1024, nullptr, 0, nullptr, v1buf, 512, nullptr, 0,
         64, 512, 1024, 0, 2);
    // scores/softmax/context
    attention_k<<<dim3(64), dim3(256), 0, stream>>>(v1buf, v2buf, Wa3, enc, ctxbf);
    // ctxp = tanh(ctx @ Wenc2dec^T), bf16 mirror for the gates GEMM
    gemm(ctxbf, 1024, wE2d, 1024, nullptr, 0, nullptr, ctxp, 512, ctx2bf, 512,
         64, 512, 1024, 1, 2);
    // gates = decX[t] + h @ Whh^T
    gemm(hDbf, 1024, wDWhh, 1024, decX + (long)t * 64 * 4096, 4096, nullptr,
         gatesD, 4096, nullptr, 0, 64, 4096, 1024, 0, 2);
    // gates += tanh(ctx) @ Wih[:,512:1024]^T   (in-place Cinit accumulate)
    gemm(ctx2bf, 512, wDWih + 512, 1024, gatesD, 4096, nullptr,
         gatesD, 4096, nullptr, 0, 64, 4096, 512, 0, 2);
    // LSTM cell -> new h (bf16 for recurrence + bf16 history for logits)
    lstm_cell_k<<<dim3(256), dim3(256), 0, stream>>>(
        gatesD, cD, nullptr, 0, hDbf, 1024,
        hhbf + (long)t * 1024, (long)TCAP_ * 1024, 1024);
  }

  // --- batched logits: [1280,1024] x [10000,1024]^T + bout ---------------
  gemm(hhbf, 1024, wO, 1024, nullptr, 0, bout, (float*)d_out, V_, nullptr, 0,
       1280, V_, 1024, 0, 4);
}
